// KNN_layer_81741817578209
// MI455X (gfx1250) — matrix operand order repaired and checked by reading, not compiled, not run on hardware
//
#include <hip/hip_runtime.h>
#include <cstdint>

typedef __attribute__((ext_vector_type(16))) _Float16 v16h;
typedef __attribute__((ext_vector_type(8)))  float    v8f;
typedef __attribute__((ext_vector_type(4)))  int      v4i;

constexpr int N      = 8192;
constexpr int D      = 2048;
constexpr int TOPK   = 16;
constexpr int BLOCK  = 256;      // 8 waves (wave32)
constexpr int WAVES  = 8;
constexpr int ROWS_PER_WG = WAVES * 16;   // 128
constexpr int JCOLS  = 64;       // column panel width (4 WMMA tiles)
constexpr int KCHUNK = 128;      // K staged per LDS refill
constexpr int NCHUNK = D / KCHUNK;        // 16
constexpr int BSTRIDE = KCHUNK + 8;       // f16 per col in LDS (pad -> conflict-free)
constexpr int STAGE_ITERS = (JCOLS * KCHUNK / 8) / BLOCK;  // 4 b128s per thread

#define NEG_INF (-__builtin_inff())

// ---- async global->LDS copy (CDNA5), with portable fallback --------------
#if defined(__has_builtin)
#if __has_builtin(__builtin_amdgcn_global_load_async_to_lds_b128)
#define HAVE_ASYNC_LDS 1
#endif
#endif

typedef __attribute__((address_space(1))) v4i* gbl_v4i_p;
typedef __attribute__((address_space(3))) v4i* lds_v4i_p;

__device__ __forceinline__ void copy_b128_to_lds(const _Float16* g, _Float16* l) {
#ifdef HAVE_ASYNC_LDS
  __builtin_amdgcn_global_load_async_to_lds_b128(
      (gbl_v4i_p)(uintptr_t)g,
      (lds_v4i_p)(uint32_t)(uintptr_t)l,
      0, 0);
#else
  *(uint4*)l = *(const uint4*)g;   // sync fallback (same barrier structure is safe)
#endif
}

#ifdef HAVE_ASYNC_LDS
#if defined(__has_builtin) && __has_builtin(__builtin_amdgcn_s_wait_asynccnt)
#define WAIT_ASYNC(n) __builtin_amdgcn_s_wait_asynccnt(n)
#else
#define WAIT_ASYNC(n) asm volatile("s_wait_asynccnt %0" :: "i"(n) : "memory")
#endif
#else
#define WAIT_ASYNC(n)
#endif

// ---- Kernel 1: row L2 normalization, f32 -> f16 --------------------------
__global__ __launch_bounds__(256) void normalize_rows(const float* __restrict__ x,
                                                      _Float16* __restrict__ xh) {
  const int row = blockIdx.x;
  const int t   = threadIdx.x;
  const float* xr = x + (size_t)row * D;
  float s = 0.f;
  for (int i = t; i < D; i += 256) { float v = xr[i]; s += v * v; }
  __shared__ float red[256];
  red[t] = s;
  __syncthreads();
  for (int off = 128; off > 0; off >>= 1) {
    if (t < off) red[t] += red[t + off];
    __syncthreads();
  }
  const float scale = 1.0f / fmaxf(sqrtf(red[0]), 1e-12f);
  _Float16* xo = xh + (size_t)row * D;
  for (int i = t; i < D; i += 256) xo[i] = (_Float16)(xr[i] * scale);
}

// ---- Kernel 2: fused WMMA Gram-matrix + per-row top-16 -------------------
__global__ __launch_bounds__(BLOCK) void knn_topk(const _Float16* __restrict__ Xh,
                                                  int* __restrict__ out) {
  __shared__ _Float16 Bbuf[2][JCOLS * BSTRIDE];   // 2 x 17408 B (double buffer)
  __shared__ float    Cbuf[WAVES][16][18];        // 9216 B (stride 18: conflict-free)

  const int tid  = threadIdx.x;
  const int wv   = tid >> 5;
  const int lane = tid & 31;
  const int lo   = lane & 15;
  const int hi   = lane >> 4;
  const int rowBase = blockIdx.x * ROWS_PER_WG + wv * 16;
  const _Float16* arow = Xh + (size_t)(rowBase + lo) * D;

  // cooperative-stage addressing (each thread: STAGE_ITERS b128 copies)
  const int scol = tid >> 4;          // 0..15 base col (16 threads per col slice? no:)
  // lin = it*BLOCK + tid ; col = lin>>4 ; sub = lin&15  (16 b128 per column)

  float vals[TOPK];
  int   idxs[TOPK];
  float minV = NEG_INF;
#pragma unroll
  for (int s = 0; s < TOPK; ++s) { vals[s] = NEG_INF; idxs[s] = 0; }
  (void)scol;

  for (int j0 = 0; j0 < N; j0 += JCOLS) {
    v8f acc[4];
#pragma unroll
    for (int jt = 0; jt < 4; ++jt) acc[jt] = (v8f){0.f,0.f,0.f,0.f,0.f,0.f,0.f,0.f};

    // prologue: stage chunk 0 into buffer 0
#pragma unroll
    for (int it = 0; it < STAGE_ITERS; ++it) {
      int lin = it * BLOCK + tid;
      int col = lin >> 4, sub = lin & 15;
      copy_b128_to_lds(Xh + (size_t)(j0 + col) * D + sub * 8,
                       &Bbuf[0][col * BSTRIDE + sub * 8]);
    }

    for (int ci = 0; ci < NCHUNK; ++ci) {
      const int cur = ci & 1;
      if (ci + 1 < NCHUNK) {
        // stage next chunk into the other buffer (overlaps with compute below)
        const int kn = (ci + 1) * KCHUNK;
#pragma unroll
        for (int it = 0; it < STAGE_ITERS; ++it) {
          int lin = it * BLOCK + tid;
          int col = lin >> 4, sub = lin & 15;
          copy_b128_to_lds(Xh + (size_t)(j0 + col) * D + kn + sub * 8,
                           &Bbuf[cur ^ 1][col * BSTRIDE + sub * 8]);
        }
        WAIT_ASYNC(STAGE_ITERS);   // chunk ci landed (in-order); next stays in flight
      } else {
        WAIT_ASYNC(0);
      }
      __syncthreads();             // all waves: data ready, prior readers done

      const _Float16* bb = &Bbuf[cur][0];
      const int kc = ci * KCHUNK;
#pragma unroll
      for (int kb = 0; kb < KCHUNK; kb += 32) {
        // A fragment (interleaved K per ISA layout): two 16B chunks
        const int kA = kc + kb + hi * 8;
        union { uint4 u[2]; v16h h; } A;
        A.u[0] = *(const uint4*)(arow + kA);
        A.u[1] = *(const uint4*)(arow + kA + 16);
        // load ALL B fragments first (distinct regs -> clause + back-to-back wmma)
        union { uint4 u[2]; v16h h; } Bv[4];
#pragma unroll
        for (int jt = 0; jt < 4; ++jt) {
          const _Float16* bp = bb + (jt * 16 + lo) * BSTRIDE + kb + hi * 8;
          Bv[jt].u[0] = *(const uint4*)bp;
          Bv[jt].u[1] = *(const uint4*)(bp + 16);
        }
#pragma unroll
        for (int jt = 0; jt < 4; ++jt)
          acc[jt] = __builtin_amdgcn_wmma_f32_16x16x32_f16(
              false, A.h, false, Bv[jt].h, (short)0, acc[jt], false, false);
      }
      __syncthreads();             // compute done before buffer is re-staged
    }

    // spill + merge one 16x16 tile at a time (wave-private Cbuf, DS in-order)
#pragma unroll
    for (int jt = 0; jt < 4; ++jt) {
#pragma unroll
      for (int r = 0; r < 8; ++r)
        Cbuf[wv][r + hi * 8][lo] = acc[jt][r];
      if (lane < 16) {
        for (int c = 0; c < 16; ++c) {
          float v = Cbuf[wv][lane][c];
          if (v > minV) {
            int done = 0;
#pragma unroll
            for (int s = 0; s < TOPK; ++s)
              if (!done && vals[s] == minV) { vals[s] = v; idxs[s] = j0 + jt * 16 + c; done = 1; }
            float m = vals[0];
#pragma unroll
            for (int s = 1; s < TOPK; ++s) m = fminf(m, vals[s]);
            minV = m;
          }
        }
      }
    }
  }

  // final: selection sort (value desc, index asc ties) via dead B buffer
  if (lane < 16) {
    float* sc = (float*)&Bbuf[0][0] + (size_t)(wv * 16 + lane) * 32;
#pragma unroll
    for (int s = 0; s < TOPK; ++s) {
      sc[s]        = vals[s];
      sc[TOPK + s] = __int_as_float(idxs[s]);
    }
    int* orow = out + (size_t)(rowBase + lane) * TOPK;
    for (int k = 0; k < TOPK; ++k) {
      float bv = NEG_INF; int bid = 0x7fffffff; int bs = 0;
      for (int s = 0; s < TOPK; ++s) {
        float v = sc[s];
        int  id = __float_as_int(sc[TOPK + s]);
        if (v > bv || (v == bv && id < bid)) { bv = v; bid = id; bs = s; }
      }
      orow[k] = bid;
      sc[bs] = NEG_INF;
    }
  }
}

extern "C" void kernel_launch(void* const* d_in, const int* in_sizes, int n_in,
                              void* d_out, int out_size, void* d_ws, size_t ws_size,
                              hipStream_t stream) {
  const float* x  = (const float*)d_in[0];
  _Float16*    Xh = (_Float16*)d_ws;          // 8192*2048*2 = 32 MB scratch
  int*         out = (int*)d_out;             // 8192*16 int32 indices

  normalize_rows<<<N, 256, 0, stream>>>(x, Xh);
  knn_topk<<<N / ROWS_PER_WG, BLOCK, 0, stream>>>(Xh, out);
}
